// HgnnEncoder_33913061769619
// MI455X (gfx1250) — matrix operand order repaired and checked by reading, not compiled, not run on hardware
//
#include <hip/hip_runtime.h>
#include <hip/hip_bf16.h>
#include <math.h>

// ---------------------------------------------------------------------------
// HGNN encoder forward for MI455X (gfx1250, wave32, WMMA).
// GEMMs: bf16 WMMA (v_wmma_f32_16x16x32_bf16) with f32 accumulation,
//        4 N-tiles per wave to amortize fp32 A reads (4 WMMA per A fragment).
// Sparse hconv: L2-resident float atomics, Binv/Dinv/bias/relu fused.
// ---------------------------------------------------------------------------

#define N_NODES  50000
#define N_HEDGES 10000
#define NNZ      400000
#define IN_CH    512
#define HID      256
#define GATE     128

typedef __attribute__((ext_vector_type(16))) __bf16 v16bf;
typedef __attribute__((ext_vector_type(8)))  float  v8f;

// ---------------------------------------------------------------------------
// Utility kernels
// ---------------------------------------------------------------------------
__global__ void zero_f(float* __restrict__ p, size_t n) {
  size_t i = (size_t)blockIdx.x * blockDim.x + threadIdx.x;
  if (i < n) p[i] = 0.f;
}

__global__ void deg_count(const int* __restrict__ nidx, const int* __restrict__ hidx,
                          float* __restrict__ degN, float* __restrict__ degE, int nnz) {
  int i = blockIdx.x * blockDim.x + threadIdx.x;
  if (i < nnz) {
    atomicAdd(&degN[nidx[i]], 1.f);
    atomicAdd(&degE[hidx[i]], 1.f);
  }
}

__global__ void inv_deg(float* __restrict__ d, int n) {
  int i = blockIdx.x * blockDim.x + threadIdx.x;
  if (i < n) { float v = d[i]; d[i] = (v > 0.f) ? (1.f / v) : 0.f; }
}

// Pack fp32 weight W[K][N] into bf16 arranged exactly as WMMA B fragments:
// linear index = ((kt*Ntiles + nt)*512) + lane*16 + e,
// with n = nt*16 + (lane&15), k = kt*32 + (e>>3)*16 + (lane>>4)*8 + (e&7).
__global__ void pack_w_bf16(const float* __restrict__ W, __bf16* __restrict__ Bp,
                            int K, int N) {
  int idx = blockIdx.x * blockDim.x + threadIdx.x;
  if (idx >= K * N) return;
  int e    = idx & 15;
  int lane = (idx >> 4) & 31;
  int tile = idx >> 9;
  int Ntiles = N >> 4;
  int nt = tile % Ntiles;
  int kt = tile / Ntiles;
  int n = (nt << 4) + (lane & 15);
  int k = (kt << 5) + ((e >> 3) << 4) + ((lane >> 4) << 3) + (e & 7);
  Bp[idx] = (__bf16)W[(size_t)k * N + n];
}

// ---------------------------------------------------------------------------
// WMMA GEMM: C[M,N] = act(A[M,K] @ W + bias).  A fp32 row-major, W pre-packed
// bf16 fragments.  Block = 8 waves; each wave computes a 16x64 strip
// (4 adjacent 16x16 tiles) so one A fragment feeds 4 WMMAs.
// ACT: 0=none, 1=relu, 2=sigmoid
// ---------------------------------------------------------------------------
template<int ACT>
__global__ __launch_bounds__(256)
void gemm_wmma_bf16(const float* __restrict__ A, const __bf16* __restrict__ Bp,
                    const float* __restrict__ bias, float* __restrict__ C,
                    int M, int N, int K) {
  const int Ntiles = N >> 4;
  const int Mtiles = M >> 4;
  const int wave = threadIdx.x >> 5;
  const int lane = threadIdx.x & 31;
  const int hl   = lane >> 4;       // half-wave: 0 or 1
  const int mr   = lane & 15;
  const int nt0  = blockIdx.x << 2; // first of 4 n-tiles for this wave
  const int mt   = blockIdx.y * 8 + wave;
  if (mt >= Mtiles) return;         // wave-uniform -> EXEC stays all-ones
  const int mbase = mt << 4;

  v8f acc[4] = {{}, {}, {}, {}};
  // A-fragment addressing (16-bit A 16x32 layout):
  //   lane holds row m=mr; element e maps to k = (e>>3)*16 + hl*8 + (e&7)
  const float*  Arow  = A + (size_t)(mbase + mr) * K + (hl << 3);
  const __bf16* Bt    = Bp + ((size_t)nt0 << 9) + ((size_t)lane << 4);
  const size_t  bstep = (size_t)Ntiles << 9;

  for (int kt = 0; kt < (K >> 5); ++kt) {
    __builtin_prefetch(Arow + 32, 0, 1);     // global_prefetch of next A chunk
    float4 f0 = *(const float4*)(Arow + 0);  // k = hl*8 + 0..3
    float4 f1 = *(const float4*)(Arow + 4);  // k = hl*8 + 4..7
    float4 f2 = *(const float4*)(Arow + 16); // k = 16 + hl*8 + 0..3
    float4 f3 = *(const float4*)(Arow + 20); // k = 16 + hl*8 + 4..7
    v16bf a;
    a[0]  = (__bf16)f0.x; a[1]  = (__bf16)f0.y; a[2]  = (__bf16)f0.z; a[3]  = (__bf16)f0.w;
    a[4]  = (__bf16)f1.x; a[5]  = (__bf16)f1.y; a[6]  = (__bf16)f1.z; a[7]  = (__bf16)f1.w;
    a[8]  = (__bf16)f2.x; a[9]  = (__bf16)f2.y; a[10] = (__bf16)f2.z; a[11] = (__bf16)f2.w;
    a[12] = (__bf16)f3.x; a[13] = (__bf16)f3.y; a[14] = (__bf16)f3.z; a[15] = (__bf16)f3.w;
    // 4 adjacent B tiles are contiguous in the packed layout
    v16bf b0 = *(const v16bf*)(Bt + 0);
    v16bf b1 = *(const v16bf*)(Bt + 512);
    v16bf b2 = *(const v16bf*)(Bt + 1024);
    v16bf b3 = *(const v16bf*)(Bt + 1536);
    acc[0] = __builtin_amdgcn_wmma_f32_16x16x32_bf16(false, a, false, b0, (short)0, acc[0], false, false);
    acc[1] = __builtin_amdgcn_wmma_f32_16x16x32_bf16(false, a, false, b1, (short)0, acc[1], false, false);
    acc[2] = __builtin_amdgcn_wmma_f32_16x16x32_bf16(false, a, false, b2, (short)0, acc[2], false, false);
    acc[3] = __builtin_amdgcn_wmma_f32_16x16x32_bf16(false, a, false, b3, (short)0, acc[3], false, false);
    Arow += 32;
    Bt   += bstep;
  }

#pragma unroll
  for (int t = 0; t < 4; ++t) {
    const int col = ((nt0 + t) << 4) + mr;
    const float bv = bias ? bias[col] : 0.f;
#pragma unroll
    for (int v = 0; v < 8; ++v) {
      // C layout: VGPR v -> M = v (lanes 0-15) / v+8 (lanes 16-31); N = lane&15
      float o = acc[t][v] + bv;
      if (ACT == 1) o = fmaxf(o, 0.f);
      if (ACT == 2) o = 1.f / (1.f + __expf(-o));
      C[(size_t)(mbase + (hl << 3) + v) * N + col] = o;
    }
  }
}

// ---------------------------------------------------------------------------
// Hypergraph conv sparse path (F = HID = 256 channels, float4 vectorized)
// ---------------------------------------------------------------------------
__global__ void scatter_edges(const float* __restrict__ xw, const int* __restrict__ nidx,
                              const int* __restrict__ hidx, float* __restrict__ eacc,
                              int nnz) {
  size_t idx = (size_t)blockIdx.x * blockDim.x + threadIdx.x;
  if (idx >= (size_t)nnz * 64) return;
  int ed = (int)(idx >> 6);
  int c4 = (int)(idx & 63) << 2;
  int src = nidx[ed], dst = hidx[ed];
  const float4 v = *(const float4*)(xw + (size_t)src * HID + c4);
  float* d = eacc + (size_t)dst * HID + c4;
  atomicAdd(d + 0, v.x); atomicAdd(d + 1, v.y);
  atomicAdd(d + 2, v.z); atomicAdd(d + 3, v.w);
}

__global__ void scatter_nodes(const float* __restrict__ eacc, const float* __restrict__ Binv,
                              const int* __restrict__ nidx, const int* __restrict__ hidx,
                              float* __restrict__ nacc, int nnz) {
  size_t idx = (size_t)blockIdx.x * blockDim.x + threadIdx.x;
  if (idx >= (size_t)nnz * 64) return;
  int ed = (int)(idx >> 6);
  int c4 = (int)(idx & 63) << 2;
  int src = hidx[ed], dst = nidx[ed];
  float s = Binv[src];
  float4 v = *(const float4*)(eacc + (size_t)src * HID + c4);
  float* d = nacc + (size_t)dst * HID + c4;
  atomicAdd(d + 0, v.x * s); atomicAdd(d + 1, v.y * s);
  atomicAdd(d + 2, v.z * s); atomicAdd(d + 3, v.w * s);
}

__global__ void hconv_fin(const float* __restrict__ nacc, const float* __restrict__ Dinv,
                          const float* __restrict__ bias, float* __restrict__ out, int rows) {
  size_t i = (size_t)blockIdx.x * blockDim.x + threadIdx.x;
  if (i >= (size_t)rows * HID) return;
  int r  = (int)(i >> 8);
  int ch = (int)(i & (HID - 1));
  float v = nacc[i] * Dinv[r] + bias[ch];
  out[i] = fmaxf(v, 0.f);
}

// ---------------------------------------------------------------------------
// BatchNorm: fused combine + per-channel stats, then normalize
// ---------------------------------------------------------------------------
__global__ __launch_bounds__(256)
void combine3_stats(const float* __restrict__ a, const float* __restrict__ b,
                    const float* __restrict__ c, float* __restrict__ outp,
                    float* __restrict__ stats, int rows) {
  int ch = threadIdx.x;            // 256 channels
  float s = 0.f, ss = 0.f;
  for (int r = blockIdx.x; r < rows; r += gridDim.x) {
    size_t i = (size_t)r * HID + ch;
    float v = a[i] + b[i] + c[i];
    outp[i] = v; s += v; ss += v * v;
  }
  atomicAdd(&stats[ch], s);
  atomicAdd(&stats[HID + ch], ss);
}

__global__ __launch_bounds__(256)
void combine2_stats(const float* __restrict__ a, const float* __restrict__ b,
                    float* __restrict__ outp, float* __restrict__ stats, int rows) {
  int ch = threadIdx.x;
  float s = 0.f, ss = 0.f;
  for (int r = blockIdx.x; r < rows; r += gridDim.x) {
    size_t i = (size_t)r * HID + ch;
    float v = 0.8f * a[i] + 0.2f * b[i];
    outp[i] = v; s += v; ss += v * v;
  }
  atomicAdd(&stats[ch], s);
  atomicAdd(&stats[HID + ch], ss);
}

__global__ void bn_apply(const float* __restrict__ t, const float* __restrict__ stats,
                         const float* __restrict__ g, const float* __restrict__ bta,
                         float* __restrict__ h, int rows) {
  size_t i = (size_t)blockIdx.x * blockDim.x + threadIdx.x;
  if (i >= (size_t)rows * HID) return;
  int ch = (int)(i & (HID - 1));
  float inv_n = 1.0f / (float)rows;
  float mean = stats[ch] * inv_n;
  float var  = stats[HID + ch] * inv_n - mean * mean;
  float is   = rsqrtf(var + 1e-5f);
  h[i] = (t[i] - mean) * is * g[ch] + bta[ch];
}

// ---------------------------------------------------------------------------
// Host orchestration.  Requires ws_size >= ~295 MB.
// ---------------------------------------------------------------------------
extern "C" void kernel_launch(void* const* d_in, const int* in_sizes, int n_in,
                              void* d_out, int out_size, void* d_ws, size_t ws_size,
                              hipStream_t stream) {
  (void)in_sizes; (void)n_in; (void)out_size; (void)ws_size;

  const float* x       = (const float*)d_in[0];
  const int*   edge    = (const int*)d_in[1];
  const int*   nidx    = edge;            // edge[0,:]
  const int*   hidx    = edge + NNZ;      // edge[1,:]
  const float* ln_w    = (const float*)d_in[2];
  const float* conv1_w = (const float*)d_in[3];
  const float* conv1_b = (const float*)d_in[4];
  const float* conv3_w = (const float*)d_in[5];
  const float* conv3_b = (const float*)d_in[6];
  const float* fc1_w   = (const float*)d_in[7];
  const float* fc1_b   = (const float*)d_in[8];
  const float* fc2_w   = (const float*)d_in[9];
  const float* fc2_b   = (const float*)d_in[10];
  const float* bn_g[4], *bn_b[4];
  for (int i = 0; i < 4; ++i) { bn_g[i] = (const float*)d_in[11 + i];
                                bn_b[i] = (const float*)d_in[15 + i]; }
  const float* convs_w[3], *convs_b[3], *f1s_w[3], *f1s_b[3], *f2s_w[3], *f2s_b[3];
  for (int i = 0; i < 3; ++i) {
    convs_w[i] = (const float*)d_in[19 + i];
    convs_b[i] = (const float*)d_in[22 + i];
    f1s_w[i]   = (const float*)d_in[25 + i];
    f1s_b[i]   = (const float*)d_in[28 + i];
    f2s_w[i]   = (const float*)d_in[31 + i];
    f2s_b[i]   = (const float*)d_in[34 + i];
  }

  float* ws = (float*)d_ws;
  const size_t NF = (size_t)N_NODES * HID;
  size_t o = 0;
  float* Dinv  = ws + o; o += N_NODES;
  float* Binv  = ws + o; o += N_HEDGES;
  float* stats = ws + o; o += 2 * HID;
  o = (o + 63) & ~(size_t)63;
  float* x0   = ws + o; o += NF;
  float* hbuf = ws + o; o += NF;
  float* tbuf = ws + o; o += NF;
  float* xw   = ws + o; o += NF;
  float* nacc = ws + o; o += NF;
  float* eacc = ws + o; o += (size_t)N_HEDGES * HID;
  float* sbuf = ws + o; o += (size_t)N_NODES * GATE;
  __bf16* pk = (__bf16*)(ws + o);
  __bf16* p_ln = pk; pk += IN_CH * HID;
  __bf16* p_c1 = pk; pk += IN_CH * HID;
  __bf16* p_c3 = pk; pk += HID * HID;
  __bf16* p_f1 = pk; pk += HID * GATE;
  __bf16* p_f2 = pk; pk += GATE * HID;
  __bf16 *p_cs[3], *p_f1s[3], *p_f2s[3];
  for (int i = 0; i < 3; ++i) {
    p_cs[i]  = pk; pk += HID * HID;
    p_f1s[i] = pk; pk += HID * GATE;
    p_f2s[i] = pk; pk += GATE * HID;
  }

  auto zero = [&](float* p, size_t n) {
    zero_f<<<dim3((unsigned)((n + 255) / 256)), dim3(256), 0, stream>>>(p, n);
  };
  auto pack = [&](const float* W, __bf16* dst, int K, int N) {
    pack_w_bf16<<<dim3((K * N + 255) / 256), dim3(256), 0, stream>>>(W, dst, K, N);
  };
  auto gemm = [&](const float* A, const __bf16* Bp, const float* bias, float* Cm,
                  int M, int N, int K, int act) {
    dim3 g((N >> 4) / 4, (unsigned)(((M >> 4) + 7) / 8));  // 4 n-tiles per wave
    if (act == 0)      gemm_wmma_bf16<0><<<g, 256, 0, stream>>>(A, Bp, bias, Cm, M, N, K);
    else if (act == 1) gemm_wmma_bf16<1><<<g, 256, 0, stream>>>(A, Bp, bias, Cm, M, N, K);
    else               gemm_wmma_bf16<2><<<g, 256, 0, stream>>>(A, Bp, bias, Cm, M, N, K);
  };
  auto hconv_post = [&](const float* xwin, const float* bias, float* outp) {
    zero(eacc, (size_t)N_HEDGES * HID);
    zero(nacc, NF);
    size_t te = (size_t)NNZ * 64;
    unsigned gb = (unsigned)((te + 255) / 256);
    scatter_edges<<<dim3(gb), 256, 0, stream>>>(xwin, nidx, hidx, eacc, NNZ);
    scatter_nodes<<<dim3(gb), 256, 0, stream>>>(eacc, Binv, nidx, hidx, nacc, NNZ);
    hconv_fin<<<dim3((unsigned)((NF + 255) / 256)), 256, 0, stream>>>(nacc, Dinv, bias, outp, N_NODES);
  };
  unsigned gNF = (unsigned)((NF + 255) / 256);

  // Degrees -> Binv / Dinv
  zero(Dinv, N_NODES); zero(Binv, N_HEDGES);
  deg_count<<<dim3((NNZ + 255) / 256), 256, 0, stream>>>(nidx, hidx, Dinv, Binv, NNZ);
  inv_deg<<<dim3((N_NODES + 255) / 256), 256, 0, stream>>>(Dinv, N_NODES);
  inv_deg<<<dim3((N_HEDGES + 255) / 256), 256, 0, stream>>>(Binv, N_HEDGES);

  // Pack all weights to bf16 fragment layout
  pack(ln_w, p_ln, IN_CH, HID);
  pack(conv1_w, p_c1, IN_CH, HID);
  pack(conv3_w, p_c3, HID, HID);
  pack(fc1_w, p_f1, HID, GATE);
  pack(fc2_w, p_f2, GATE, HID);
  for (int i = 0; i < 3; ++i) {
    pack(convs_w[i], p_cs[i], HID, HID);
    pack(f1s_w[i], p_f1s[i], HID, GATE);
    pack(f2s_w[i], p_f2s[i], GATE, HID);
  }

  // x0 = x @ ln_w ; x1 = relu(hconv(x, conv1))
  gemm(x, p_ln, nullptr, x0, N_NODES, HID, IN_CH, 0);
  gemm(x, p_c1, nullptr, xw, N_NODES, HID, IN_CH, 0);
  hconv_post(xw, conv1_b, hbuf);                               // hbuf = x1
  // SE gate: s1 = sigmoid(relu(x1@fc1+b)@fc2+b)
  gemm(hbuf, p_f1, fc1_b, sbuf, N_NODES, GATE, HID, 1);
  gemm(sbuf, p_f2, fc2_b, tbuf, N_NODES, HID, GATE, 2);        // tbuf = s1
  // h = BN(x1 + s1 + x0)
  zero(stats, 2 * HID);
  combine3_stats<<<dim3(512), dim3(256), 0, stream>>>(hbuf, tbuf, x0, xw, stats, N_NODES);
  bn_apply<<<dim3(gNF), 256, 0, stream>>>(xw, stats, bn_g[0], bn_b[0], hbuf, N_NODES);

  for (int i = 0; i < 3; ++i) {
    gemm(hbuf, p_cs[i], nullptr, xw, N_NODES, HID, HID, 0);
    hconv_post(xw, convs_b[i], tbuf);                          // tbuf = relu(hconv)
    zero(stats, 2 * HID);
    combine2_stats<<<dim3(512), dim3(256), 0, stream>>>(tbuf, x0, xw, stats, N_NODES);
    bn_apply<<<dim3(gNF), 256, 0, stream>>>(xw, stats, bn_g[i + 1], bn_b[i + 1], hbuf, N_NODES);
    gemm(hbuf, p_f1s[i], f1s_b[i], sbuf, N_NODES, GATE, HID, 1);
    gemm(sbuf, p_f2s[i], f2s_b[i], hbuf, N_NODES, HID, GATE, 2); // h replaced (as in ref)
  }

  // out = relu(hconv(h, conv3))
  gemm(hbuf, p_c3, nullptr, xw, N_NODES, HID, HID, 0);
  hconv_post(xw, conv3_b, (float*)d_out);
}
